// NonLocalBlock_5600637354631
// MI455X (gfx1250) — matrix operand order repaired and checked by reading, not compiled, not run on hardware
//
#include <hip/hip_runtime.h>

// ---------------------------------------------------------------------------
// NonLocalBlock (GroupNorm + QKV 1x1conv + softmax attention + residual)
// MI455X / gfx1250: wave32, v_wmma_f32_16x16x32_bf16, fp32 accumulate.
// Two-pass (split-softmax) attention. B-side matrices pre-packed in WMMA
// B-fragment order (fragment = two b128 loads/lane). K tiles staged through
// LDS via async global->LDS b128 loads (ASYNCcnt) with plain-copy fallback.
// ---------------------------------------------------------------------------

typedef __attribute__((ext_vector_type(16))) __bf16 v16bf;
typedef __attribute__((ext_vector_type(8)))  __bf16 v8bf;
typedef __attribute__((ext_vector_type(8)))  float  v8f;

constexpr int B_  = 4;
constexpr int C_  = 512;
constexpr int HW_ = 4096;   // 64*64
constexpr int G_  = 32;
constexpr int CG_ = 16;     // C_/G_
// logits scaled into exp2 domain: s2 = s * (1/sqrt(512)) * log2(e)
constexpr float SCL2_ = 0.044194173824159216f * 1.4426950408889634f;

#if __has_builtin(__builtin_amdgcn_global_load_async_to_lds_b128) && \
    __has_builtin(__builtin_amdgcn_s_wait_asynccnt)
#define ASYNC_LDS_COPY 1
typedef int v4i_ __attribute__((vector_size(4 * sizeof(int))));
typedef __attribute__((address_space(1))) v4i_ gv4i;  // global (AS1)
typedef __attribute__((address_space(3))) v4i_ lv4i;  // LDS (AS3)
#endif

// Copy `n16` 16-byte chunks global -> LDS, strided by the block's 256 threads.
__device__ __forceinline__ void stage_to_lds(uint4* dst, const uint4* src,
                                             int n16, int tid) {
#ifdef ASYNC_LDS_COPY
  for (int t = tid; t < n16; t += 256)
    __builtin_amdgcn_global_load_async_to_lds_b128((gv4i*)(src + t),
                                                   (lv4i*)(dst + t), 0, 0);
  __builtin_amdgcn_s_wait_asynccnt(0);
#else
  for (int t = tid; t < n16; t += 256) dst[t] = src[t];
#endif
}

__device__ __forceinline__ __bf16 to_bf16(float f) {
  unsigned u = __builtin_bit_cast(unsigned, f);
  unsigned short h = (unsigned short)((u + 0x7FFFu + ((u >> 16) & 1u)) >> 16); // RNE
  return __builtin_bit_cast(__bf16, h);
}

__device__ __forceinline__ v8f vzero8() {
  v8f z = {0.f, 0.f, 0.f, 0.f, 0.f, 0.f, 0.f, 0.f};
  return z;
}

__device__ __forceinline__ v8f wmma_bf16(v16bf a, v16bf b, v8f c) {
  return __builtin_amdgcn_wmma_f32_16x16x32_bf16(false, a, false, b, (short)0, c,
                                                 false, false);
}

// Packed-B element index for element (k, n) of a [K][N] B-matrix.
// tile(nt = n/16, kc = k/32) stored at (nt*ktiles + kc)*512; within the tile
// lane L = (n%16) + 16*((k%32)/16) owns elements e = k%16 (32 contiguous B).
__device__ __forceinline__ size_t packed_idx(int k, int n, int ktiles) {
  return ((size_t)(n >> 4) * ktiles + (k >> 5)) * 512 +
         (size_t)(((n & 15) | (((k >> 4) & 1) << 4)) << 4) + (k & 15);
}

// B-fragment (32x16) from a packed tile base: two contiguous b128 loads.
__device__ __forceinline__ v16bf load_b_packed(const __bf16* tile, int lane) {
  const __bf16* p = tile + (lane << 4);
  v8bf lo = *(const v8bf*)p;
  v8bf hi = *(const v8bf*)(p + 8);
  v16bf r;
#pragma unroll
  for (int e = 0; e < 8; ++e) { r[e] = lo[e]; r[e + 8] = hi[e]; }
  return r;
}

// A-fragment (16x32) from row-major rows (lda in elements): two b128 loads.
__device__ __forceinline__ v16bf load_a_frag(const __bf16* A, int lda, int m0,
                                             int k0, int lane) {
  const __bf16* p = A + (size_t)(m0 + (lane & 15)) * lda + k0 + ((lane >> 4) << 4);
  v8bf lo = *(const v8bf*)p;
  v8bf hi = *(const v8bf*)(p + 8);
  v16bf r;
#pragma unroll
  for (int e = 0; e < 8; ++e) { r[e] = lo[e]; r[e + 8] = hi[e]; }
  return r;
}

// ---------------------------------------------------------------------------
// Kernel 1: weight prep — Wq,Wv -> packed-B [c][o]; Wk -> row-major (A use)
// ---------------------------------------------------------------------------
__global__ void prep_weights_k(const float* __restrict__ qw,
                               const float* __restrict__ kw,
                               const float* __restrict__ vw,
                               __bf16* __restrict__ wqP, __bf16* __restrict__ wkN,
                               __bf16* __restrict__ wvP) {
  int idx = blockIdx.x * blockDim.x + threadIdx.x;
  if (idx < C_ * C_) {
    int o = idx / C_, c = idx % C_;
    size_t pk = packed_idx(c, o, 16);  // B[c][o], K=512 -> 16 k-tiles
    wqP[pk] = to_bf16(qw[idx]);
    wvP[pk] = to_bf16(vw[idx]);
    wkN[idx] = to_bf16(kw[idx]);
  }
}

// ---------------------------------------------------------------------------
// Kernel 2: GroupNorm -> hP (packed-B [c][i] per batch) and hT ([b][i][c])
// ---------------------------------------------------------------------------
__global__ void groupnorm_k(const float* __restrict__ x,
                            const float* __restrict__ gw,
                            const float* __restrict__ gb,
                            __bf16* __restrict__ hP, __bf16* __restrict__ hT) {
  __shared__ float s1[256], s2[256];
  const int NE = CG_ * HW_;  // 65536
  int b = blockIdx.x / G_, g = blockIdx.x % G_;
  int tid = threadIdx.x;
  size_t base = ((size_t)b * C_ + g * CG_) * HW_;

  float sum = 0.f, sq = 0.f;
  for (int e = tid; e < NE; e += 256) {
    float v = x[base + e];
    sum += v; sq += v * v;
  }
  s1[tid] = sum; s2[tid] = sq;
  __syncthreads();
  for (int s = 128; s > 0; s >>= 1) {
    if (tid < s) { s1[tid] += s1[tid + s]; s2[tid] += s2[tid + s]; }
    __syncthreads();
  }
  float mu = s1[0] * (1.0f / NE);
  float var = s2[0] * (1.0f / NE) - mu * mu;
  float rstd = rsqrtf(var + 1e-6f);

  for (int e = tid; e < NE; e += 256) {
    int cl = e >> 12;
    int i  = e & (HW_ - 1);
    int c  = g * CG_ + cl;
    float v = (x[base + e] - mu) * rstd * gw[c] + gb[c];
    __bf16 bv = to_bf16(v);
    hP[(size_t)b * (C_ * HW_) + packed_idx(c, i, 16)] = bv;
    hT[((size_t)b * HW_ + i) * C_ + c] = bv;
  }
}

// ---------------------------------------------------------------------------
// Kernel 3: bf16 GEMM  D[M][N] = A[M][512] * Bpacked[512][N] + bias
// 32x64 strip per wave (2 A-frags x 4 B-frags -> 8 WMMAs per k-step,
// ~22 FLOP/B from L2); optional packed-B-format output.
// ---------------------------------------------------------------------------
__global__ void gemm_bf16_k(const __bf16* __restrict__ A,
                            const __bf16* __restrict__ Bp,
                            const float* __restrict__ bias,
                            __bf16* __restrict__ D,
                            int M, int N, int lda, int ldd,
                            long aStride, long bStride, long dStride,
                            int bias_on_m, int packed_out) {
  int z = blockIdx.z;
  A  += (size_t)z * aStride;
  Bp += (size_t)z * bStride;
  D  += (size_t)z * dStride;

  int lane = threadIdx.x & 31;
  int wave = (blockIdx.x * blockDim.x + threadIdx.x) >> 5;
  int strips_n = N >> 6;
  int sm = wave / strips_n, sn = wave % strips_n;
  if (sm >= (M >> 5)) return;
  int m0 = sm << 5, n0 = sn << 6;

  const __bf16* bbase = Bp + (size_t)(n0 >> 4) * (16 * 512);
  v8f acc[2][4];
#pragma unroll
  for (int mi = 0; mi < 2; ++mi)
#pragma unroll
    for (int ni = 0; ni < 4; ++ni) acc[mi][ni] = vzero8();

#pragma unroll 2
  for (int k0 = 0; k0 < 512; k0 += 32) {
    v16bf a0 = load_a_frag(A, lda, m0,      k0, lane);
    v16bf a1 = load_a_frag(A, lda, m0 + 16, k0, lane);
#pragma unroll
    for (int ni = 0; ni < 4; ++ni) {
      v16bf b = load_b_packed(bbase + ((size_t)ni * 16 + (k0 >> 5)) * 512, lane);
      acc[0][ni] = wmma_bf16(a0, b, acc[0][ni]);
      acc[1][ni] = wmma_bf16(a1, b, acc[1][ni]);
    }
  }

#pragma unroll
  for (int mi = 0; mi < 2; ++mi) {
    int tmx = (m0 >> 4) + mi;
    int mr  = (tmx << 4) + ((lane >> 4) << 3);
#pragma unroll
    for (int ni = 0; ni < 4; ++ni) {
      int n = n0 + ni * 16 + (lane & 15);
      if (packed_out) {
        size_t pidx = ((size_t)((n0 >> 4) + ni) * (M >> 5) + (tmx >> 1)) * 512 +
                      (size_t)(((lane & 15) | ((tmx & 1) << 4)) << 4) +
                      ((lane >> 4) << 3);
        v8bf st;
#pragma unroll
        for (int r = 0; r < 8; ++r)
          st[r] = to_bf16(acc[mi][ni][r] + (bias_on_m ? bias[mr + r] : bias[n]));
        *(v8bf*)(D + pidx) = st;
      } else {
#pragma unroll
        for (int r = 0; r < 8; ++r) {
          float v = acc[mi][ni][r] + (bias_on_m ? bias[mr + r] : bias[n]);
          D[(size_t)(mr + r) * ldd + n] = to_bf16(v);
        }
      }
    }
  }
}

// ---------------------------------------------------------------------------
// Kernel 4: attention pass A — per-row max m2[b,i] (exp2 domain) and sumexp2.
// 8 waves/block, wave owns a 16-row i-tile; K j-tiles staged through LDS.
// ---------------------------------------------------------------------------
__global__ void attn_stats_k(const __bf16* __restrict__ qT,
                             const __bf16* __restrict__ kP,
                             float* __restrict__ mbuf, float* __restrict__ lbuf) {
  __shared__ __attribute__((aligned(16))) __bf16 Ks[16 * 512];  // 16KB j-tile
  int b = blockIdx.z;
  int tid = threadIdx.x;
  int lane = tid & 31;
  int w = tid >> 5;
  int i0 = blockIdx.x * 128 + w * 16;
  const __bf16* qb = qT + (size_t)b * HW_ * C_;        // row-major [i][c]
  const __bf16* kb = kP + (size_t)b * C_ * HW_;        // packed tiles

  v16bf qa[16];
#pragma unroll
  for (int kk = 0; kk < 16; ++kk) qa[kk] = load_a_frag(qb, C_, i0, kk * 32, lane);

  float mx[8], sm[8];
#pragma unroll
  for (int r = 0; r < 8; ++r) { mx[r] = -3.0e38f; sm[r] = 0.f; }

  for (int j0 = 0; j0 < HW_; j0 += 16) {
    int jt = j0 >> 4;
    stage_to_lds((uint4*)Ks, (const uint4*)(kb + (size_t)jt * (16 * 512)), 1024,
                 tid);
    if (j0 + 16 < HW_)
      __builtin_prefetch(kb + (size_t)(jt + 1) * (16 * 512) + tid * 32, 0, 1);
    __syncthreads();

    v8f acc = vzero8();
#pragma unroll
    for (int kk = 0; kk < 16; ++kk) {
      v16bf bfr = load_b_packed(Ks + (kk << 9), lane);  // ds_load_b128 x2
      acc = wmma_bf16(qa[kk], bfr, acc);
    }
    __syncthreads();  // before next stage overwrites Ks

#pragma unroll
    for (int r = 0; r < 8; ++r) {
      float s = acc[r] * SCL2_;
      float nm = fmaxf(mx[r], s);
      sm[r] = sm[r] * exp2f(mx[r] - nm) + exp2f(s - nm);
      mx[r] = nm;
    }
  }
  // merge 16 j-lanes within each 16-lane half (online-softmax combine)
#pragma unroll
  for (int off = 8; off > 0; off >>= 1) {
#pragma unroll
    for (int r = 0; r < 8; ++r) {
      float om = __shfl_xor(mx[r], off, 32);
      float os = __shfl_xor(sm[r], off, 32);
      float nm = fmaxf(mx[r], om);
      sm[r] = sm[r] * exp2f(mx[r] - nm) + os * exp2f(om - nm);
      mx[r] = nm;
    }
  }
  if ((lane & 15) == 0) {
#pragma unroll
    for (int r = 0; r < 8; ++r) {
      int i = i0 + r + ((lane >> 4) << 3);
      mbuf[(size_t)b * HW_ + i] = mx[r];
      lbuf[(size_t)b * HW_ + i] = sm[r];
    }
  }
}

// ---------------------------------------------------------------------------
// Kernel 5: attention pass B — out = x + (1/l) * sum_j 2^(S-m) vT[j][c].
// Block: 32 i-rows, 8 waves, jc-chunks of 64. Q rows + K j-block in LDS;
// P repacked through LDS (C-frag -> A-frag transpose); V from packed L2.
// ---------------------------------------------------------------------------
__global__ void attn_out_k(const float* __restrict__ x,
                           const __bf16* __restrict__ qT,
                           const __bf16* __restrict__ kP,
                           const __bf16* __restrict__ vP,
                           const float* __restrict__ mbuf,
                           const float* __restrict__ lbuf,
                           float* __restrict__ out) {
  __shared__ __attribute__((aligned(16))) __bf16 Qs[32 * 512];  // 32KB
  __shared__ __attribute__((aligned(16))) __bf16 Ks[64 * 512];  // 64KB (4 j-tiles)
  __shared__ __attribute__((aligned(16))) __bf16 Ps[32 * 64];   // 4KB

  int b = blockIdx.z;
  int i_block = blockIdx.x * 32;
  int tid = threadIdx.x;
  int lane = tid & 31;
  int w = tid >> 5;

  const __bf16* qb = qT + ((size_t)b * HW_ + i_block) * C_;
  const __bf16* kb = kP + (size_t)b * C_ * HW_;
  const __bf16* vb = vP + (size_t)b * 128 * 512;  // batch offset in packed V

  stage_to_lds((uint4*)Qs, (const uint4*)qb, 2048, tid);

  int it_s = w & 1, jt_s = w >> 1;        // S phase: i-tile, j-subtile(0..3)
  int it_o = w & 1, cb = (w >> 1) * 128;  // O phase: i-tile, c-block base

  float mrow[8], linv[8];
#pragma unroll
  for (int r = 0; r < 8; ++r) {
    int hs = r + ((lane >> 4) << 3);
    mrow[r] = mbuf[(size_t)b * HW_ + i_block + it_s * 16 + hs];
    linv[r] = 1.0f / lbuf[(size_t)b * HW_ + i_block + it_o * 16 + hs];
  }

  v8f o[8];
#pragma unroll
  for (int ct = 0; ct < 8; ++ct) o[ct] = vzero8();

  for (int jc = 0; jc < HW_; jc += 64) {
    // stage 4 j-tiles of K (contiguous 64KB in packed layout)
    stage_to_lds((uint4*)Ks, (const uint4*)(kb + (size_t)(jc >> 4) * (16 * 512)),
                 4096, tid);
    if (jc + 64 < HW_)
      __builtin_prefetch(kb + (size_t)((jc >> 4) + 4) * (16 * 512) + tid * 128,
                         0, 1);
    __syncthreads();  // Ks (and first-iter Qs) ready; prior iter done (Ps safe)

    // ---- S tile (16 i x 16 j), K=c=512, K-frags from LDS ----
    v8f acc = vzero8();
#pragma unroll
    for (int kk = 0; kk < 16; ++kk) {
      v16bf a   = load_a_frag(Qs, C_, it_s * 16, kk * 32, lane);
      v16bf bfr = load_b_packed(Ks + (((jt_s * 16) + kk) << 9), lane);
      acc = wmma_bf16(a, bfr, acc);
    }
    int prow = it_s * 16 + ((lane >> 4) << 3);
    int pcol = jt_s * 16 + (lane & 15);
#pragma unroll
    for (int r = 0; r < 8; ++r) {
      float p = exp2f(acc[r] * SCL2_ - mrow[r]);
      Ps[(prow + r) * 64 + pcol] = to_bf16(p);
    }
    __syncthreads();  // Ps ready

    // ---- O[16 i x 128 c] += P(16x32) * V(32x128), V packed from L2 ----
#pragma unroll
    for (int jj = 0; jj < 64; jj += 32) {
      v16bf a = load_a_frag(Ps, 64, it_o * 16, jj, lane);
#pragma unroll
      for (int ct = 0; ct < 8; ++ct) {
        const __bf16* vtile =
            vb + ((size_t)((cb >> 4) + ct) * 512 + ((jc + jj) >> 5)) * 512;
        o[ct] = wmma_bf16(a, load_b_packed(vtile, lane), o[ct]);
      }
    }
    __syncthreads();  // all P reads done before next Ks/Ps overwrite
  }

  // epilogue: out[b][c][i] = x[b][c][i] + O[i][c] / l[i]
  int irow = i_block + it_o * 16 + ((lane >> 4) << 3);
#pragma unroll
  for (int ct = 0; ct < 8; ++ct) {
    int c = cb + ct * 16 + (lane & 15);
#pragma unroll
    for (int r = 0; r < 8; ++r) {
      size_t idx = ((size_t)b * C_ + c) * HW_ + irow + r;
      out[idx] = x[idx] + o[ct][r] * linv[r];
    }
  }
}

// ---------------------------------------------------------------------------
// launch
// ---------------------------------------------------------------------------
extern "C" void kernel_launch(void* const* d_in, const int* in_sizes, int n_in,
                              void* d_out, int out_size, void* d_ws, size_t ws_size,
                              hipStream_t stream) {
  (void)in_sizes; (void)n_in; (void)out_size; (void)ws_size;
  const float* x    = (const float*)d_in[0];
  const float* gn_w = (const float*)d_in[1];
  const float* gn_b = (const float*)d_in[2];
  const float* q_w  = (const float*)d_in[3];
  const float* q_b  = (const float*)d_in[4];
  const float* k_w  = (const float*)d_in[5];
  const float* k_b  = (const float*)d_in[6];
  const float* v_w  = (const float*)d_in[7];
  const float* v_b  = (const float*)d_in[8];
  float* out = (float*)d_out;

  char* ws = (char*)d_ws;
  const size_t SB = (size_t)B_ * HW_ * C_ * 2;  // 16 MiB per bf16 activation buf
  __bf16* hT  = (__bf16*)(ws + 0 * SB);  // [b][i][c] row-major (A use)
  __bf16* hP  = (__bf16*)(ws + 1 * SB);  // packed-B [c][i] per batch
  __bf16* qT  = (__bf16*)(ws + 2 * SB);  // [b][i][c] row-major (A use)
  __bf16* kPb = (__bf16*)(ws + 3 * SB);  // packed-B [c][j] per batch
  __bf16* vPb = (__bf16*)(ws + 4 * SB);  // packed-B [j][c], flattened batches
  __bf16* wqP = (__bf16*)(ws + 5 * SB);
  __bf16* wvP = (__bf16*)(ws + 5 * SB + (512u << 10));
  __bf16* wkN = (__bf16*)(ws + 5 * SB + (1024u << 10));
  float*  mb  = (float*) (ws + 5 * SB + (1536u << 10));
  float*  lb  = (float*) (ws + 5 * SB + (1536u << 10) + (64u << 10));

  prep_weights_k<<<dim3((C_ * C_ + 255) / 256), dim3(256), 0, stream>>>(
      q_w, k_w, v_w, wqP, wkN, wvP);
  groupnorm_k<<<dim3(B_ * G_), dim3(256), 0, stream>>>(x, gn_w, gn_b, hP, hT);

  // qT[i][o] = hT[i][c] * WqP   -> row-major (used as A in attention)
  gemm_bf16_k<<<dim3((B_ * HW_ / 32) * (C_ / 64) / 8, 1, 1), dim3(256), 0, stream>>>(
      hT, wqP, q_b, qT, B_ * HW_, C_, C_, C_, 0, 0, 0, 0, 0);
  // vP[j][c] = hT * WvP        -> packed-B (used as B in O-phase)
  gemm_bf16_k<<<dim3((B_ * HW_ / 32) * (C_ / 64) / 8, 1, 1), dim3(256), 0, stream>>>(
      hT, wvP, v_b, vPb, B_ * HW_, C_, C_, 0, 0, 0, 0, 0, 1);
  // kP[c][j] = Wk * hP (per batch) -> packed-B (used as B in S-phase)
  gemm_bf16_k<<<dim3((C_ / 32) * (HW_ / 64) / 8, 1, B_), dim3(256), 0, stream>>>(
      wkN, hP, k_b, kPb, C_, HW_, C_, 0,
      0, (long)C_ * HW_, (long)C_ * HW_, 1, 1);

  attn_stats_k<<<dim3(HW_ / 128, 1, B_), dim3(256), 0, stream>>>(qT, kPb, mb, lb);
  attn_out_k<<<dim3(HW_ / 32, 1, B_), dim3(256), 0, stream>>>(
      x, qT, kPb, vPb, mb, lb, out);
}
